// MoE_layer_flux_46712064311453
// MI455X (gfx1250) — matrix operand order repaired and checked by reading, not compiled, not run on hardware
//
#include <hip/hip_runtime.h>
#include <hip/hip_bf16.h>

typedef __attribute__((ext_vector_type(16))) __bf16 bf16x16;
typedef __attribute__((ext_vector_type(8)))  __bf16 bf16x8;
typedef __attribute__((ext_vector_type(4)))  __bf16 bf16x4;
typedef __attribute__((ext_vector_type(2)))  __bf16 bf16x2;
typedef __attribute__((ext_vector_type(8)))  float  v8f;

constexpr int T_ = 2048, H_ = 1024, F_ = 4096, E_ = 8, K_ = 2, M_ = T_ * K_;
constexpr int MAXT = 40;          // max 128-row expert tiles: sum ceil(c_e/128) <= 39

// ---------------- cast x -> bf16 ----------------
__global__ __launch_bounds__(256) void cast_x_kernel(const float* __restrict__ x,
                                                     __bf16* __restrict__ xb) {
  int i = (blockIdx.x * 256 + threadIdx.x) * 4;
  float4 f = *(const float4*)(x + i);
  bf16x4 o;
  o[0] = (__bf16)f.x; o[1] = (__bf16)f.y; o[2] = (__bf16)f.z; o[3] = (__bf16)f.w;
  *(bf16x4*)(xb + i) = o;
}

// ---------------- router: logits, top-2, softmax ----------------
__global__ __launch_bounds__(256) void router_kernel(const float* __restrict__ x,
                                                     const float* __restrict__ rw,
                                                     int* __restrict__ tidx,
                                                     float* __restrict__ probs) {
  int wid = threadIdx.x >> 5, lane = threadIdx.x & 31;
  int t = blockIdx.x * 8 + wid;
  const float* xr = x + (size_t)t * H_;
  float a[E_] = {};
  for (int j = lane; j < H_; j += 32) {
    float xv = xr[j];
    const float* r = rw + j * E_;
#pragma unroll
    for (int e = 0; e < E_; e++) a[e] += xv * r[e];
  }
#pragma unroll
  for (int s = 16; s > 0; s >>= 1)
#pragma unroll
    for (int e = 0; e < E_; e++) a[e] += __shfl_xor(a[e], s, 32);
  if (lane == 0) {
    int i0 = 0; float b0 = a[0];
    for (int e = 1; e < E_; e++) if (a[e] > b0) { b0 = a[e]; i0 = e; }
    int i1 = -1; float b1 = -3.0e38f;
    for (int e = 0; e < E_; e++) if (e != i0 && a[e] > b1) { b1 = a[e]; i1 = e; }
    float e0 = 1.0f, e1 = __expf(b1 - b0);
    float inv = 1.0f / (e0 + e1);
    tidx[t * 2] = i0;  tidx[t * 2 + 1] = i1;
    probs[t * 2] = e0 * inv;  probs[t * 2 + 1] = e1 * inv;
  }
}

// ---------------- stable counting sort by expert + tile tables ----------------
__global__ __launch_bounds__(256) void scan_kernel(const int* __restrict__ tidx,
                                                   int* __restrict__ rowsrc,
                                                   int* __restrict__ posm,
                                                   int* __restrict__ tile_e,
                                                   int* __restrict__ tile_r0,
                                                   int* __restrict__ tile_end,
                                                   int* __restrict__ ntiles) {
  __shared__ int cnt[256][E_];
  __shared__ int base[E_ + 1];
  int tid = threadIdx.x;
  int m0 = tid * 16;
  int es[16];
  int lc[E_] = {};
#pragma unroll
  for (int i = 0; i < 16; i++) { int e = tidx[m0 + i]; es[i] = e; lc[e]++; }
#pragma unroll
  for (int e = 0; e < E_; e++) cnt[tid][e] = lc[e];
  __syncthreads();
  if (tid == 0) {
    int s = 0;
    for (int e = 0; e < E_; e++) {
      base[e] = s;
      int tt = 0;
      for (int j = 0; j < 256; j++) tt += cnt[j][e];
      s += tt;
    }
    base[E_] = s;
  }
  __syncthreads();
  int off[E_];
#pragma unroll
  for (int e = 0; e < E_; e++) {
    int s = base[e];
    for (int j = 0; j < tid; j++) s += cnt[j][e];
    off[e] = s;
  }
#pragma unroll
  for (int i = 0; i < 16; i++) {
    int e = es[i];
    int p = off[e]++;
    int m = m0 + i;
    rowsrc[p] = m >> 1;   // token id of this grouped slot
    posm[m]   = p;        // inverse permutation
  }
  if (tid == 0) {
    int nt = 0;
    for (int e = 0; e < E_; e++) {
      int b = base[e], c = base[e + 1] - base[e];
      for (int j = 0; j * 128 < c; j++) {
        tile_e[nt] = e; tile_r0[nt] = b + j * 128; tile_end[nt] = b + c; nt++;
      }
    }
    *ntiles = nt;
  }
}

// ---------------- grouped GEMM engine (bf16 WMMA, f32 accum, double-buffered) ----
// GEMM==1: gelu(x_g @ w1[e]) -> f32 out + bf16 shadow.  GEMM==2: gelu_g @ w2[e] -> f32 y.
template <int GEMM, int KD, int ND>
__global__ __launch_bounds__(256) void gemm_kernel(const __bf16* __restrict__ A,
                                                   const float* __restrict__ Wall,
                                                   const int* __restrict__ rowsrc,
                                                   const int* __restrict__ tile_e,
                                                   const int* __restrict__ tile_r0,
                                                   const int* __restrict__ tile_end,
                                                   const int* __restrict__ ntiles,
                                                   float* __restrict__ outF,
                                                   __bf16* __restrict__ outB) {
  constexpr int NCOL = ND / 128;
  int t = blockIdx.x / NCOL, nb = blockIdx.x % NCOL;
  if (t >= *ntiles) return;
  int e = tile_e[t], r0 = tile_r0[t], rend = tile_end[t];
  const float* W = Wall + (size_t)e * KD * ND;
  int n0 = nb * 128;

  // double-buffered tiles; 40-elem row stride = 80B, 16B-aligned, bank-spread
  __shared__ __bf16 lA[2][128][40];
  __shared__ __bf16 lB[2][128][40];   // N-major: lB[buf][n][k]

  int tid = threadIdx.x, lane = tid & 31, wid = tid >> 5;
  int wm = wid & 3, wn = wid >> 2;

  v8f acc[2][4] = {};

  // A staging: 128 rows x 32 k, two async b128 per thread
  int lrow = tid & 127;
  int lcol = (tid >> 7) * 16;
  int r = r0 + lrow;
  int rr = (r < rend) ? r : (rend - 1);
  const __bf16* ap = (GEMM == 1) ? (A + (size_t)rowsrc[rr] * KD)
                                 : (A + (size_t)rr * KD);
  // B staging: 32 k x 128 n f32 -> packed bf16 pairs along K, 16 f32 per thread
  int nq = (tid & 31) * 4;
  int kp = (tid >> 5) * 2;   // 0,2,...,14

  auto load_tile = [&](int kt, int buf) {
    // --- A: global -> LDS direct (ASYNCcnt path) ---
    unsigned la = (unsigned)(size_t)&lA[buf][lrow][lcol];
    unsigned long long ga = (unsigned long long)(size_t)(ap + kt + lcol);
    asm volatile("global_load_async_to_lds_b128 %0, %1, off"
                 :: "v"(la), "v"(ga) : "memory");
    asm volatile("global_load_async_to_lds_b128 %0, %1, off offset:16"
                 :: "v"(la), "v"(ga) : "memory");
    // --- B: f32 -> packed bf16 (k,k+1) pairs, b32 LDS stores ---
#pragma unroll
    for (int kk = kp; kk < 32; kk += 16) {
      const float* w0 = W + (size_t)(kt + kk) * ND + n0 + nq;
      float4 fa = *(const float4*)w0;
      float4 fb = *(const float4*)(w0 + ND);
      bf16x2 p;
      p[0] = (__bf16)fa.x; p[1] = (__bf16)fb.x; *(bf16x2*)&lB[buf][nq + 0][kk] = p;
      p[0] = (__bf16)fa.y; p[1] = (__bf16)fb.y; *(bf16x2*)&lB[buf][nq + 1][kk] = p;
      p[0] = (__bf16)fa.z; p[1] = (__bf16)fb.z; *(bf16x2*)&lB[buf][nq + 2][kk] = p;
      p[0] = (__bf16)fa.w; p[1] = (__bf16)fb.w; *(bf16x2*)&lB[buf][nq + 3][kk] = p;
    }
  };

  load_tile(0, 0);

  for (int k0 = 0; k0 < KD; k0 += 32) {
    int buf = (k0 >> 5) & 1;
    asm volatile("s_wait_asynccnt 0x0" ::: "memory");
    __syncthreads();
    if (k0 + 32 < KD) load_tile(k0 + 32, buf ^ 1);

    // A frag: lane<16 -> M=lane, K 0-7 / 16-23 ; lane>=16 -> K 8-15 / 24-31
    bf16x16 af[2], bfr[4];
    int mrow0 = wm * 32 + (lane & 15);
    int ah = (lane < 16) ? 0 : 8;
#pragma unroll
    for (int mi = 0; mi < 2; mi++) {
      bf16x8 lo = *(const bf16x8*)&lA[buf][mrow0 + mi * 16][ah];
      bf16x8 hi = *(const bf16x8*)&lA[buf][mrow0 + mi * 16][16 + ah];
      af[mi] = __builtin_shufflevector(lo, hi, 0, 1, 2, 3, 4, 5, 6, 7, 8, 9, 10, 11, 12, 13, 14, 15);
    }
    // B frag: N=lane%16; lane<16 -> K 0-7 / 16-23, lane>=16 -> K 8-15 / 24-31
    int nrow0 = wn * 64 + (lane & 15);
    int bh = (lane < 16) ? 0 : 8;
#pragma unroll
    for (int ni = 0; ni < 4; ni++) {
      bf16x8 lo = *(const bf16x8*)&lB[buf][nrow0 + ni * 16][bh];
      bf16x8 hi = *(const bf16x8*)&lB[buf][nrow0 + ni * 16][bh + 16];
      bfr[ni] = __builtin_shufflevector(lo, hi, 0, 1, 2, 3, 4, 5, 6, 7, 8, 9, 10, 11, 12, 13, 14, 15);
    }
#pragma unroll
    for (int mi = 0; mi < 2; mi++)
#pragma unroll
      for (int ni = 0; ni < 4; ni++)
        acc[mi][ni] = __builtin_amdgcn_wmma_f32_16x16x32_bf16(
            false, af[mi], false, bfr[ni], (short)0, acc[mi][ni], false, false);
  }

  // epilogue: C/D layout — VGPR i: M = i (+8 for upper lane half), N = lane%16
  int colb = n0 + wn * 64 + (lane & 15);
  int rowb = r0 + wm * 32 + ((lane < 16) ? 0 : 8);
#pragma unroll
  for (int mi = 0; mi < 2; mi++)
#pragma unroll
    for (int ni = 0; ni < 4; ni++)
#pragma unroll
      for (int i = 0; i < 8; i++) {
        int rg = rowb + mi * 16 + i;
        if (rg < rend) {
          float v = acc[mi][ni][i];
          size_t o = (size_t)rg * ND + colb + ni * 16;
          if constexpr (GEMM == 1) {
            float g = 0.5f * v * (1.0f + erff(v * 0.70710678118654752f));
            outF[o] = g;
            outB[o] = (__bf16)g;
          } else {
            outF[o] = v;
          }
        }
      }
}

// ---------------- weighted combine (deterministic, no atomics) ----------------
__global__ __launch_bounds__(256) void combine_kernel(const float* __restrict__ y,
                                                      const float* __restrict__ probs,
                                                      const int* __restrict__ posm,
                                                      float* __restrict__ out) {
  int idx = blockIdx.x * 256 + threadIdx.x;
  int t = idx >> 10;             // H_ = 1024
  int h = idx & 1023;
  int p0 = posm[2 * t], p1 = posm[2 * t + 1];
  out[idx] = probs[2 * t]     * y[(size_t)p0 * H_ + h] +
             probs[2 * t + 1] * y[(size_t)p1 * H_ + h];
}

extern "C" void kernel_launch(void* const* d_in, const int* in_sizes, int n_in,
                              void* d_out, int out_size, void* d_ws, size_t ws_size,
                              hipStream_t stream) {
  const float* x  = (const float*)d_in[0];
  const float* rw = (const float*)d_in[1];
  const float* w1 = (const float*)d_in[2];
  const float* w2 = (const float*)d_in[3];

  float* out_mlp  = (float*)d_out;
  float* out_gelu = out_mlp + (size_t)T_ * H_;

  char* ws = (char*)d_ws;
  size_t off = 0;
  auto take = [&](size_t bytes) { size_t r = off; off += (bytes + 255) & ~(size_t)255; return r; };
  __bf16* xb    = (__bf16*)(ws + take((size_t)T_ * H_ * 2));
  __bf16* gelub = (__bf16*)(ws + take((size_t)M_ * F_ * 2));
  float*  ybuf  = (float*)(ws + take((size_t)M_ * H_ * 4));
  float*  probs = (float*)(ws + take((size_t)T_ * 2 * 4));
  int* tidx     = (int*)(ws + take((size_t)T_ * 2 * 4));
  int* rowsrc   = (int*)(ws + take((size_t)M_ * 4));
  int* posm     = (int*)(ws + take((size_t)M_ * 4));
  int* tile_e   = (int*)(ws + take(64 * 4));
  int* tile_r0  = (int*)(ws + take(64 * 4));
  int* tile_end = (int*)(ws + take(64 * 4));
  int* ntiles   = (int*)(ws + take(16));

  cast_x_kernel<<<(T_ * H_) / 1024, 256, 0, stream>>>(x, xb);
  router_kernel<<<T_ / 8, 256, 0, stream>>>(x, rw, tidx, probs);
  scan_kernel<<<1, 256, 0, stream>>>(tidx, rowsrc, posm, tile_e, tile_r0, tile_end, ntiles);
  gemm_kernel<1, H_, F_><<<MAXT * (F_ / 128), 256, 0, stream>>>(
      xb, w1, rowsrc, tile_e, tile_r0, tile_end, ntiles, out_gelu, gelub);
  gemm_kernel<2, F_, H_><<<MAXT * (H_ / 128), 256, 0, stream>>>(
      gelub, w2, rowsrc, tile_e, tile_r0, tile_end, ntiles, ybuf, nullptr);
  combine_kernel<<<(T_ * H_) / 256, 256, 0, stream>>>(ybuf, probs, posm, out_mlp);
}
